// MultiHeadAttention_71485435675030
// MI455X (gfx1250) — compile-verified, matrix operand order checked by reading, not run on hardware
//
#include <hip/hip_runtime.h>

// MHA: B=4, S=2048, E=1024, H=16, D=64 — f16 WMMA flash attention + TDM staging.
#define BB 4
#define SS 2048
#define EE 1024
#define HH 16
#define DD 64

typedef __attribute__((ext_vector_type(16))) _Float16 v16h;
typedef __attribute__((ext_vector_type(8)))  float    v8f;
typedef __attribute__((ext_vector_type(4)))  unsigned v4u;
typedef __attribute__((ext_vector_type(8)))  int      v8i;
typedef __attribute__((ext_vector_type(4)))  int      v4i;

union FragH { v16h h; v4u u[2]; };

__device__ __forceinline__ v8f wmma16(v16h a, v16h b, v8f c) {
  return __builtin_amdgcn_wmma_f32_16x16x32_f16(false, a, false, b, (short)0, c, false, false);
}

// ---------------------------------------------------------------------------
// Tensor Data Mover: 2D tile load (2-byte elements) global -> LDS.
// D# layout per CDNA5 ISA §8.3 (group0) / §8.4 (group1); groups 2/3 unused (2D).
//   group0: count=1 | lds_addr | global_addr[31:0] | global_addr[56:32]+type=2
//   group1: data_size=2B | tensor_dim0/1 | tile_dim0/1 | tensor_dim0_stride
// Tile rows land contiguously in LDS: row*tile_d0 elements apart.
// This toolchain's builtin takes 6 args (groups 2/3 + trailing group zeroed).
// ---------------------------------------------------------------------------
__device__ __forceinline__ void tdm_load_2d_f16(
    unsigned lds_off, const _Float16* gptr,
    unsigned tensor_d0, unsigned tensor_d1,
    unsigned tile_d0, unsigned tile_d1,
    unsigned long long stride0_elems) {
  const unsigned long long ga = (unsigned long long)(size_t)gptr;
  v4u g0;
  g0[0] = 1u;                                    // count=1, user-mode, no gather
  g0[1] = lds_off;                               // LDS byte address
  g0[2] = (unsigned)(ga & 0xFFFFFFFFu);          // global_addr[31:0]
  g0[3] = (unsigned)((ga >> 32) & 0x1FFFFFFu) | (2u << 30);  // [56:32] | type=2
  v8i g1;
  g1[0] = (int)(1u << 16);                       // workgroup_mask=0, data_size=1 (2B)
  g1[1] = (int)((tensor_d0 & 0xFFFFu) << 16);    // tensor_dim0[15:0] @ bits63:48
  g1[2] = (int)(((tensor_d0 >> 16) & 0xFFFFu) | ((tensor_d1 & 0xFFFFu) << 16));
  g1[3] = (int)(((tensor_d1 >> 16) & 0xFFFFu) | ((tile_d0 & 0xFFFFu) << 16));
  g1[4] = (int)(tile_d1 & 0xFFFFu);              // tile_dim1; tile_dim2=0
  g1[5] = (int)(unsigned)(stride0_elems & 0xFFFFFFFFu);
  g1[6] = (int)(unsigned)((stride0_elems >> 32) & 0xFFFFu);  // dim1_stride=0 (2D)
  g1[7] = 0;
  const v4i zero4 = {0, 0, 0, 0};
  const v8i zero8 = {0, 0, 0, 0, 0, 0, 0, 0};
  __builtin_amdgcn_tensor_load_to_lds(g0, g1, zero4, zero4, zero8, 0);
}

__device__ __forceinline__ unsigned lds_offset_of(const void* p) {
  return (unsigned)(size_t)p;  // generic LDS pointer: low 32 bits = LDS address
}

// ---------------------------------------------------------------------------
// Weight transpose + fp32->f16 convert: Wt[out][in] = (f16) W[in][out]
// ---------------------------------------------------------------------------
__global__ __launch_bounds__(256) void wt_cvt_kernel(const float* __restrict__ w,
                                                     _Float16* __restrict__ wt) {
  __shared__ float tile[32][33];
  const int tx = threadIdx.x & 31;
  const int ty = threadIdx.x >> 5;
  const int bx = blockIdx.x;
  const int by = blockIdx.y;
  const int x = bx * 32 + tx;
#pragma unroll
  for (int j = 0; j < 4; ++j) {
    const int y = by * 32 + ty + j * 8;
    tile[ty + j * 8][tx] = w[(size_t)y * EE + x];
  }
  __syncthreads();
  const int xo = by * 32 + tx;
#pragma unroll
  for (int j = 0; j < 4; ++j) {
    const int yo = bx * 32 + ty + j * 8;
    wt[(size_t)yo * EE + xo] = (_Float16)tile[tx][ty + j * 8];
  }
}

// ---------------------------------------------------------------------------
// QKV projection. grid.z = {0:Q, 1:K, 2:V}. Block tile 128x128, K-chunk 32.
// 8 waves, each 32x64 (8 WMMA accumulators). B panel staged via TDM.
// Q/K written [b,h,s,d]; V written transposed [b,h,d,s].
// ---------------------------------------------------------------------------
__global__ __launch_bounds__(256) void qkv_proj_kernel(
    const float* __restrict__ xq, const float* __restrict__ xk, const float* __restrict__ xv,
    const _Float16* __restrict__ wtq, const _Float16* __restrict__ wtk, const _Float16* __restrict__ wtv,
    const float* __restrict__ bq, const float* __restrict__ bk, const float* __restrict__ bv,
    _Float16* __restrict__ Qh, _Float16* __restrict__ Kh, _Float16* __restrict__ Vt) {
  const int z = blockIdx.z;
  const float*    __restrict__ x    = (z == 0) ? xq  : (z == 1) ? xk  : xv;
  const _Float16* __restrict__ wt   = (z == 0) ? wtq : (z == 1) ? wtk : wtv;
  const float*    __restrict__ bias = (z == 0) ? bq  : (z == 1) ? bk  : bv;

  __shared__ alignas(16) _Float16 At[128 * 32];
  __shared__ alignas(16) _Float16 Bt[128 * 32];

  const int tid  = threadIdx.x;
  const int wvid = tid >> 5;
  const int lane = tid & 31;
  const int hf   = lane >> 4;
  const int lm   = lane & 15;
  const int wm = wvid & 3;    // 4 m-subtiles of 32
  const int wn = wvid >> 2;   // 2 n-subtiles of 64
  const int m0 = blockIdx.y * 128;
  const int n0 = blockIdx.x * 128;

  const int lrow = tid >> 1;        // 0..127 staging row
  const int lcol = (tid & 1) * 16;  // 0 or 16

  v8f acc[2][4] = {};

  const unsigned bt_off = lds_offset_of(&Bt[0]);

  for (int k0 = 0; k0 < EE; k0 += 32) {
    {  // stage A: fp32 activations -> f16 LDS (16 elems/thread)
      const float* src = x + (size_t)(m0 + lrow) * EE + k0 + lcol;
      const float4 f0 = *(const float4*)(src + 0);
      const float4 f1 = *(const float4*)(src + 4);
      const float4 f2 = *(const float4*)(src + 8);
      const float4 f3 = *(const float4*)(src + 12);
      union { _Float16 h[8]; v4u u; } t0, t1;
      t0.h[0] = (_Float16)f0.x; t0.h[1] = (_Float16)f0.y;
      t0.h[2] = (_Float16)f0.z; t0.h[3] = (_Float16)f0.w;
      t0.h[4] = (_Float16)f1.x; t0.h[5] = (_Float16)f1.y;
      t0.h[6] = (_Float16)f1.z; t0.h[7] = (_Float16)f1.w;
      t1.h[0] = (_Float16)f2.x; t1.h[1] = (_Float16)f2.y;
      t1.h[2] = (_Float16)f2.z; t1.h[3] = (_Float16)f2.w;
      t1.h[4] = (_Float16)f3.x; t1.h[5] = (_Float16)f3.y;
      t1.h[6] = (_Float16)f3.z; t1.h[7] = (_Float16)f3.w;
      *(v4u*)&At[lrow * 32 + lcol]     = t0.u;
      *(v4u*)&At[lrow * 32 + lcol + 8] = t1.u;
    }
    if (wvid == 0) {  // stage B panel (128 rows x 32 k) via Tensor Data Mover
      tdm_load_2d_f16(bt_off, wt + (size_t)n0 * EE + k0,
                      EE, EE, 32, 128, EE);
      __builtin_amdgcn_s_wait_tensorcnt((short)0);
    }
    __syncthreads();

    FragH a[2], bf[4];
#pragma unroll
    for (int i = 0; i < 2; ++i) {
      const _Float16* arow = &At[(wm * 32 + i * 16 + lm) * 32];
      a[i].u[0] = *(const v4u*)(arow + hf * 8);
      a[i].u[1] = *(const v4u*)(arow + 16 + hf * 8);
    }
#pragma unroll
    for (int j = 0; j < 4; ++j) {
      const _Float16* brow = &Bt[(wn * 64 + j * 16 + lm) * 32];
      bf[j].u[0] = *(const v4u*)(brow + hf * 8);
      bf[j].u[1] = *(const v4u*)(brow + 16 + hf * 8);
    }
#pragma unroll
    for (int i = 0; i < 2; ++i)
#pragma unroll
      for (int j = 0; j < 4; ++j)
        acc[i][j] = wmma16(a[i].h, bf[j].h, acc[i][j]);
    __syncthreads();
  }

#pragma unroll
  for (int i = 0; i < 2; ++i)
#pragma unroll
    for (int j = 0; j < 4; ++j) {
      const int col = n0 + wn * 64 + j * 16 + lm;
      const float bv2 = bias[col];
      const int hh = col >> 6, dd = col & 63;
#pragma unroll
      for (int r = 0; r < 8; ++r) {
        const int row = m0 + wm * 32 + i * 16 + r + hf * 8;
        const int bb = row >> 11, ss = row & (SS - 1);
        const float val = acc[i][j][r] + bv2;
        const size_t bh = (size_t)(bb * HH + hh);
        if (z == 2)      Vt[(bh * DD + dd) * SS + ss] = (_Float16)val;
        else if (z == 0) Qh[(bh * SS + ss) * DD + dd] = (_Float16)val;
        else             Kh[(bh * SS + ss) * DD + dd] = (_Float16)val;
      }
    }
}

// ---------------------------------------------------------------------------
// Flash attention: 1 wave = 16 query rows, key chunks of 32, online softmax.
// K/V chunks staged once per block via TDM (8x dedup vs per-wave loads).
// ---------------------------------------------------------------------------
__global__ __launch_bounds__(256) void flash_attn_kernel(
    const _Float16* __restrict__ Qh, const _Float16* __restrict__ Kh,
    const _Float16* __restrict__ Vt, const int* __restrict__ mask,
    _Float16* __restrict__ ctx) {
  __shared__ alignas(16) _Float16 Kt[32 * 64];   // [key][d]
  __shared__ alignas(16) _Float16 Vts[64 * 32];  // [d][key]
  __shared__ alignas(16) _Float16 pbuf[8][16 * 32];

  const int tid  = threadIdx.x;
  const int wvid = tid >> 5;
  const int lane = tid & 31;
  const int hf   = lane >> 4;
  const int lm   = lane & 15;
  const int b = blockIdx.z, hh = blockIdx.y;
  const int q0 = blockIdx.x * 128 + wvid * 16;
  const size_t bh = (size_t)(b * HH + hh);

  const _Float16* qrow = Qh + (bh * SS + q0 + lm) * DD;
  FragH qa0, qa1;
  qa0.u[0] = *(const v4u*)(qrow + hf * 8);
  qa0.u[1] = *(const v4u*)(qrow + 16 + hf * 8);
  qa1.u[0] = *(const v4u*)(qrow + 32 + hf * 8);
  qa1.u[1] = *(const v4u*)(qrow + 48 + hf * 8);

  const _Float16* kbase = Kh + bh * (size_t)SS * DD;
  const _Float16* vbase = Vt + bh * (size_t)DD * SS;
  const int* mrow = mask + b * SS;
  const unsigned kt_off = lds_offset_of(&Kt[0]);
  const unsigned vt_off = lds_offset_of(&Vts[0]);

  v8f o0 = {}, o1 = {}, o2 = {}, o3 = {};
  float mr[8], lr[8];
#pragma unroll
  for (int r = 0; r < 8; ++r) { mr[r] = -3.0e38f; lr[r] = 0.0f; }

  _Float16* pb = pbuf[wvid];

  for (int kk = 0; kk < SS; kk += 32) {
    if (wvid == 0) {  // TDM: K chunk (32 keys x 64 d), V chunk (64 d x 32 keys)
      tdm_load_2d_f16(kt_off, kbase + (size_t)kk * DD, DD, SS, 64, 32, DD);
      tdm_load_2d_f16(vt_off, vbase + kk, SS, DD, 32, 64, SS);
      __builtin_amdgcn_s_wait_tensorcnt((short)0);
    }
    __syncthreads();

    FragH k00, k01, k10, k11;
    {
      const _Float16* kr0 = &Kt[lm * 64];
      k00.u[0] = *(const v4u*)(kr0 + hf * 8);
      k00.u[1] = *(const v4u*)(kr0 + 16 + hf * 8);
      k01.u[0] = *(const v4u*)(kr0 + 32 + hf * 8);
      k01.u[1] = *(const v4u*)(kr0 + 48 + hf * 8);
      const _Float16* kr1 = &Kt[(16 + lm) * 64];
      k10.u[0] = *(const v4u*)(kr1 + hf * 8);
      k10.u[1] = *(const v4u*)(kr1 + 16 + hf * 8);
      k11.u[0] = *(const v4u*)(kr1 + 32 + hf * 8);
      k11.u[1] = *(const v4u*)(kr1 + 48 + hf * 8);
    }
    v8f s0 = {}, s1 = {};
    s0 = wmma16(qa0.h, k00.h, s0);
    s0 = wmma16(qa1.h, k01.h, s0);
    s1 = wmma16(qa0.h, k10.h, s1);
    s1 = wmma16(qa1.h, k11.h, s1);

    const int msk0 = mrow[kk + lm];
    const int msk1 = mrow[kk + 16 + lm];

    float corr[8];
#pragma unroll
    for (int r = 0; r < 8; ++r) {
      // mask BEFORE scale (reference order), scale = 1/sqrt(64)
      const float a0 = (msk0 ? s0[r] : -1.0e20f) * 0.125f;
      const float a1 = (msk1 ? s1[r] : -1.0e20f) * 0.125f;
      float mx = fmaxf(a0, a1);
      mx = fmaxf(mx, __shfl_xor(mx, 1, 32));
      mx = fmaxf(mx, __shfl_xor(mx, 2, 32));
      mx = fmaxf(mx, __shfl_xor(mx, 4, 32));
      mx = fmaxf(mx, __shfl_xor(mx, 8, 32));
      const float mnew = fmaxf(mr[r], mx);
      const float p0 = __expf(a0 - mnew);
      const float p1 = __expf(a1 - mnew);
      float sum = p0 + p1;
      sum += __shfl_xor(sum, 1, 32);
      sum += __shfl_xor(sum, 2, 32);
      sum += __shfl_xor(sum, 4, 32);
      sum += __shfl_xor(sum, 8, 32);
      corr[r] = __expf(mr[r] - mnew);
      lr[r] = lr[r] * corr[r] + sum;
      mr[r] = mnew;
      s0[r] = p0;
      s1[r] = p1;
    }
#pragma unroll
    for (int r = 0; r < 8; ++r) {
      o0[r] *= corr[r]; o1[r] *= corr[r]; o2[r] *= corr[r]; o3[r] *= corr[r];
    }

    // Re-layout P (C-layout -> A-layout) through wave-private LDS
#pragma unroll
    for (int r = 0; r < 8; ++r) {
      const int m = r + hf * 8;
      pb[m * 32 + lm]      = (_Float16)s0[r];
      pb[m * 32 + 16 + lm] = (_Float16)s1[r];
    }
    asm volatile("s_wait_dscnt 0x0" ::: "memory");

    FragH pa;
    const _Float16* prow = pb + lm * 32;
    pa.u[0] = *(const v4u*)(prow + hf * 8);
    pa.u[1] = *(const v4u*)(prow + 16 + hf * 8);

    FragH vf;
    const _Float16* vr0 = &Vts[(0 * 16 + lm) * 32];
    vf.u[0] = *(const v4u*)(vr0 + hf * 8);
    vf.u[1] = *(const v4u*)(vr0 + 16 + hf * 8);
    o0 = wmma16(pa.h, vf.h, o0);
    const _Float16* vr1 = &Vts[(1 * 16 + lm) * 32];
    vf.u[0] = *(const v4u*)(vr1 + hf * 8);
    vf.u[1] = *(const v4u*)(vr1 + 16 + hf * 8);
    o1 = wmma16(pa.h, vf.h, o1);
    const _Float16* vr2 = &Vts[(2 * 16 + lm) * 32];
    vf.u[0] = *(const v4u*)(vr2 + hf * 8);
    vf.u[1] = *(const v4u*)(vr2 + 16 + hf * 8);
    o2 = wmma16(pa.h, vf.h, o2);
    const _Float16* vr3 = &Vts[(3 * 16 + lm) * 32];
    vf.u[0] = *(const v4u*)(vr3 + hf * 8);
    vf.u[1] = *(const v4u*)(vr3 + 16 + hf * 8);
    o3 = wmma16(pa.h, vf.h, o3);

    __syncthreads();  // all waves done with Kt/Vts before next TDM overwrite
  }

#pragma unroll
  for (int r = 0; r < 8; ++r) {
    const int q = q0 + r + hf * 8;
    const float inv = 1.0f / lr[r];
    _Float16* dst = ctx + ((size_t)b * SS + q) * EE + hh * DD + lm;
    dst[0]  = (_Float16)(o0[r] * inv);
    dst[16] = (_Float16)(o1[r] * inv);
    dst[32] = (_Float16)(o2[r] * inv);
    dst[48] = (_Float16)(o3[r] * inv);
  }
}

// ---------------------------------------------------------------------------
// Output projection: ctx(f16) @ Wo + bo -> fp32. Both panels staged via TDM.
// Block tile 128x128, 8 waves x (32x64).
// ---------------------------------------------------------------------------
__global__ __launch_bounds__(256) void out_proj_kernel(
    const _Float16* __restrict__ ctx, const _Float16* __restrict__ wto,
    const float* __restrict__ bo, float* __restrict__ out) {
  __shared__ alignas(16) _Float16 At[128 * 32];
  __shared__ alignas(16) _Float16 Bt[128 * 32];

  const int tid  = threadIdx.x;
  const int wvid = tid >> 5;
  const int lane = tid & 31;
  const int hf   = lane >> 4;
  const int lm   = lane & 15;
  const int wm = wvid & 3;
  const int wn = wvid >> 2;
  const int m0 = blockIdx.y * 128;
  const int n0 = blockIdx.x * 128;
  const unsigned at_off = lds_offset_of(&At[0]);
  const unsigned bt_off = lds_offset_of(&Bt[0]);

  v8f acc[2][4] = {};

  for (int k0 = 0; k0 < EE; k0 += 32) {
    if (wvid == 0) {
      tdm_load_2d_f16(at_off, ctx + (size_t)m0 * EE + k0, EE, (unsigned)(BB * SS),
                      32, 128, EE);
      tdm_load_2d_f16(bt_off, wto + (size_t)n0 * EE + k0, EE, EE, 32, 128, EE);
      __builtin_amdgcn_s_wait_tensorcnt((short)0);
    }
    __syncthreads();

    FragH a[2], bf[4];
#pragma unroll
    for (int i = 0; i < 2; ++i) {
      const _Float16* arow = &At[(wm * 32 + i * 16 + lm) * 32];
      a[i].u[0] = *(const v4u*)(arow + hf * 8);
      a[i].u[1] = *(const v4u*)(arow + 16 + hf * 8);
    }
#pragma unroll
    for (int j = 0; j < 4; ++j) {
      const _Float16* brow = &Bt[(wn * 64 + j * 16 + lm) * 32];
      bf[j].u[0] = *(const v4u*)(brow + hf * 8);
      bf[j].u[1] = *(const v4u*)(brow + 16 + hf * 8);
    }
#pragma unroll
    for (int i = 0; i < 2; ++i)
#pragma unroll
      for (int j = 0; j < 4; ++j)
        acc[i][j] = wmma16(a[i].h, bf[j].h, acc[i][j]);
    __syncthreads();
  }

#pragma unroll
  for (int i = 0; i < 2; ++i)
#pragma unroll
    for (int j = 0; j < 4; ++j) {
      const int col = n0 + wn * 64 + j * 16 + lm;
      const float bias = bo[col];
#pragma unroll
      for (int r = 0; r < 8; ++r) {
        const int row = m0 + wm * 32 + i * 16 + r + hf * 8;
        out[(size_t)row * EE + col] = acc[i][j][r] + bias;
      }
    }
}

// ---------------------------------------------------------------------------
extern "C" void kernel_launch(void* const* d_in, const int* in_sizes, int n_in,
                              void* d_out, int out_size, void* d_ws, size_t ws_size,
                              hipStream_t stream) {
  const float* q    = (const float*)d_in[0];
  const float* k    = (const float*)d_in[1];
  const float* v    = (const float*)d_in[2];
  const int*   mask = (const int*)d_in[3];
  const float* wq = (const float*)d_in[4];
  const float* bq = (const float*)d_in[5];
  const float* wk = (const float*)d_in[6];
  const float* bk = (const float*)d_in[7];
  const float* wv = (const float*)d_in[8];
  const float* bv = (const float*)d_in[9];
  const float* wo = (const float*)d_in[10];
  const float* bo = (const float*)d_in[11];

  char* ws = (char*)d_ws;
  size_t off = 0;
  _Float16* wtq = (_Float16*)(ws + off); off += (size_t)EE * EE * 2;
  _Float16* wtk = (_Float16*)(ws + off); off += (size_t)EE * EE * 2;
  _Float16* wtv = (_Float16*)(ws + off); off += (size_t)EE * EE * 2;
  _Float16* wto = (_Float16*)(ws + off); off += (size_t)EE * EE * 2;
  _Float16* Qh  = (_Float16*)(ws + off); off += (size_t)BB * SS * EE * 2;
  _Float16* Kh  = (_Float16*)(ws + off); off += (size_t)BB * SS * EE * 2;
  _Float16* Vt  = (_Float16*)(ws + off); off += (size_t)BB * SS * EE * 2;
  _Float16* ctx = (_Float16*)(ws + off); off += (size_t)BB * SS * EE * 2;

  const dim3 blk(256);

  const dim3 gT(EE / 32, EE / 32);
  wt_cvt_kernel<<<gT, blk, 0, stream>>>(wq, wtq);
  wt_cvt_kernel<<<gT, blk, 0, stream>>>(wk, wtk);
  wt_cvt_kernel<<<gT, blk, 0, stream>>>(wv, wtv);
  wt_cvt_kernel<<<gT, blk, 0, stream>>>(wo, wto);

  qkv_proj_kernel<<<dim3(EE / 128, (BB * SS) / 128, 3), blk, 0, stream>>>(
      q, k, v, wtq, wtk, wtv, bq, bk, bv, Qh, Kh, Vt);

  flash_attn_kernel<<<dim3(SS / 128, HH, BB), blk, 0, stream>>>(
      Qh, Kh, Vt, mask, ctx);

  out_proj_kernel<<<dim3(EE / 128, (BB * SS) / 128, 1), blk, 0, stream>>>(
      ctx, wto, bo, (float*)d_out);
}
